// DualBranchMoENet_45372034515272
// MI455X (gfx1250) — matrix operand order repaired and checked by reading, not compiled, not run on hardware
//
#include <hip/hip_runtime.h>
#include <math.h>

typedef __bf16 bf16;
typedef __attribute__((ext_vector_type(16))) __bf16 v16bf;
typedef __attribute__((ext_vector_type(8)))  __bf16 v8bf;
typedef __attribute__((ext_vector_type(8)))  float  v8f;

#define PI_F 3.14159265358979f
#define NB   128   // batch
#define NFR  33    // STFT frames
#define NBIN 129   // rfft bins

__device__ __forceinline__ float sigf(float x){ return 1.0f/(1.0f+__expf(-x)); }

// ---------------- STFT: Hann window + 256-pt rDFT magnitude -----------------
__global__ void dbm_stft(const float* __restrict__ x, float* __restrict__ magf,
                         bf16* __restrict__ magb){
  __shared__ float s[256];
  int blk = blockIdx.x;
  int b = blk / NFR, t = blk % NFR;
  int n = threadIdx.x;
  int j = t*128 + n - 128;              // reflect pad by 128
  if (j < 0) j = -j;
  if (j > 4095) j = 2*4095 - j;
  float win = 0.5f*(1.0f - __cosf(2.0f*PI_F*(float)n/256.0f));
  s[n] = x[(size_t)b*4096 + j]*win;
  __syncthreads();
  if (n < NBIN){
    float re = 0.f, im = 0.f;
    float w = -2.0f*PI_F*(float)n/256.0f;
    for (int i = 0; i < 256; ++i){
      float sv, cv; __sincosf(w*(float)i, &sv, &cv);
      re += s[i]*cv; im += s[i]*sv;
    }
    float m = sqrtf(re*re + im*im);
    size_t o = (size_t)(b*NFR + t)*NBIN + n;
    magf[o] = m; magb[o] = (bf16)m;
  }
}

// ---------------- pooled = mag.mean(frames) ----------------
__global__ void dbm_pool_mean(const float* __restrict__ magf, float* __restrict__ pooled){
  int idx = blockIdx.x*blockDim.x + threadIdx.x;
  if (idx >= NB*NBIN) return;
  int b = idx / NBIN, k = idx % NBIN;
  float s = 0.f;
  for (int t = 0; t < NFR; ++t) s += magf[(size_t)(b*NFR + t)*NBIN + k];
  pooled[idx] = s / (float)NFR;
}

// ---------------- gate MLP + top2 + softmax ----------------
__global__ void dbm_gate(const float* __restrict__ pooled,
                         const float* __restrict__ w1, const float* __restrict__ b1,
                         const float* __restrict__ w2, const float* __restrict__ b2,
                         float* __restrict__ gw, int* __restrict__ topi){
  __shared__ float hid[128];
  __shared__ float lg[8];
  int b = blockIdx.x, j = threadIdx.x;      // 128 threads
  float a = b1[j];
  for (int k = 0; k < NBIN; ++k) a += pooled[b*NBIN + k]*w1[j*NBIN + k];
  hid[j] = fmaxf(a, 0.f);
  __syncthreads();
  if (j < 8){
    float l = b2[j];
    for (int k = 0; k < 128; ++k) l += hid[k]*w2[j*128 + k];
    lg[j] = l;
  }
  __syncthreads();
  if (j == 0){
    int i0 = 0;
    for (int e = 1; e < 8; ++e) if (lg[e] > lg[i0]) i0 = e;
    int i1 = -1;
    for (int e = 0; e < 8; ++e){ if (e == i0) continue; if (i1 < 0 || lg[e] > lg[i1]) i1 = e; }
    float e1 = __expf(lg[i1] - lg[i0]);
    gw[b*2]   = 1.f/(1.f + e1);
    gw[b*2+1] = e1/(1.f + e1);
    topi[b*2] = i0; topi[b*2+1] = i1;
  }
}

// ---- pack W (N x K row-major fp32) into WMMA-fragment-ordered bf16 B ----
// Bpk[((kblk*(N/16) + nb)*32 + lane)*16 + j] = Wt[k][n], n = nb*16 + lane%16,
// k = kblk*32 + (lane/16)*16 + j (zero-padded K). Each lane's B fragment is
// 32 contiguous bytes -> two global_load_b128.
__global__ void dbm_pack_wB(const float* __restrict__ W, bf16* __restrict__ Bpk,
                            int N, int K, int Kpad){
  int idx = blockIdx.x*blockDim.x + threadIdx.x;
  if (idx >= Kpad*N) return;
  int j    = idx & 15;
  int lane = (idx >> 4) & 31;
  int rest = idx >> 9;
  int nb16 = N >> 4;
  int nb   = rest % nb16;
  int kblk = rest / nb16;
  int n = nb*16 + (lane & 15);
  int k = kblk*32 + (lane >> 4)*16 + j;
  Bpk[idx] = (k < K) ? (bf16)W[(size_t)n*K + k] : (bf16)0.0f;
}

// ---------------- generic bf16 WMMA GEMM ----------------
// C[M,N] = act( A*B + bias + addend ).
// MODE 0: A plain MxK row-major bf16 (staged via b128 load + b128 LDS store).
// MODE 1: implicit im2col over BTC tensor in[(b*T+t)*C + c] with branchless
//         clamped loads (no exec-mask branches). T/C/KW/PAD compile-time.
// Block = 128 threads (4 waves), tile 32x128, K-step 32, double-buffered LDS.
// Per wave per K-step: 2 ds_load_b128 (A frag) + 8 global_load_b128 (B frags)
// + 4 v_wmma.
template<int MODE, int T, int C, int KW, int PAD>
__global__ __launch_bounds__(128)
void dbm_gemm(const bf16* __restrict__ A, const bf16* __restrict__ Bpk,
              const float* __restrict__ bias, const float* __restrict__ addend,
              float* __restrict__ Cf, bf16* __restrict__ Cb,
              int M, int N, int K, int relu){
  __shared__ __align__(16) bf16 As[2][32*32];
  const int tid  = threadIdx.x;
  const int lane = tid & 31;
  const int wave = tid >> 5;
  const int m0 = blockIdx.x * 32;
  const int n0 = blockIdx.y * 128;
  const int wm = (wave & 1) * 16;
  const int wn = (wave >> 1) * 64;

  // --- A staging coords (K-invariant) ---
  const int ar1 = tid >> 5;            // MODE 1: base row, +4 per sub-step
  const int ac1 = tid & 31;            // MODE 1: k offset
  const int ar0 = tid >> 2;            // MODE 0: row (32 rows x 4 chunks)
  const int ak0 = (tid & 3) * 8;       // MODE 0: 8-element (16B) chunk
  int bS[8], tS[8];
  const bf16* a0row = nullptr;
  if constexpr (MODE == 1){
    #pragma unroll
    for (int s = 0; s < 8; ++s){
      int gm = m0 + ar1 + 4*s;
      bS[s] = gm / T; tS[s] = gm - bS[s]*T;
    }
  } else {
    a0row = A + (size_t)(m0 + ar0)*K + ak0;
  }

  // --- fragment coords (K-invariant) ---
  const int frow = wm + (lane & 15);
  const int half = lane >> 4;
  const int nb16 = N >> 4;
  const int nb0  = (n0 + wn) >> 4;

  v8f acc0 = {}; v8f acc1 = {}; v8f acc2 = {}; v8f acc3 = {};
  const int KB = K >> 5;
  #pragma unroll 2
  for (int kb = 0; kb < KB; ++kb){
    const int k0 = kb << 5;
    bf16* Asw = As[kb & 1];
    if constexpr (MODE == 0){
      v8bf v = *(const v8bf*)(a0row + k0);
      *(v8bf*)&Asw[ar0*32 + ak0] = v;
    } else {
      #pragma unroll
      for (int s = 0; s < 8; ++s){
        int gk = k0 + ac1;
        int ci = gk / KW, kk = gk - ci*KW;
        int tt = tS[s] + kk - PAD;
        int ok  = (ci < C) & (tt >= 0) & (tt < T);
        int cic = (ci < C) ? ci : (C - 1);
        int ttc = (tt < 0) ? 0 : ((tt >= T) ? (T - 1) : tt);
        bf16 raw = A[(size_t)(bS[s]*T + ttc)*C + cic];   // always in-bounds
        Asw[(ar1 + 4*s)*32 + ac1] = ok ? raw : (bf16)0.0f;
      }
    }
    __syncthreads();   // single barrier per step (double-buffered LDS)

    // A fragment: two contiguous 16B chunks -> ds_load_b128 x2
    const bf16* fr = &Asw[frow*32 + half*16];
    v8bf alo = *(const v8bf*)fr;
    v8bf ahi = *(const v8bf*)(fr + 8);
    v16bf af;
    #pragma unroll
    for (int i2 = 0; i2 < 8; ++i2){ af[i2] = alo[i2]; af[8 + i2] = ahi[i2]; }

    // B fragments: 32B contiguous per lane per 16-col block (L2-resident)
    size_t boff = ((size_t)(kb*nb16 + nb0)*32 + lane)*16;
    v16bf b0 = *(const v16bf*)(Bpk + boff);
    v16bf b1 = *(const v16bf*)(Bpk + boff + 512);
    v16bf b2 = *(const v16bf*)(Bpk + boff + 1024);
    v16bf b3 = *(const v16bf*)(Bpk + boff + 1536);

    acc0 = __builtin_amdgcn_wmma_f32_16x16x32_bf16(false, af, false, b0, (short)0, acc0, false, false);
    acc1 = __builtin_amdgcn_wmma_f32_16x16x32_bf16(false, af, false, b1, (short)0, acc1, false, false);
    acc2 = __builtin_amdgcn_wmma_f32_16x16x32_bf16(false, af, false, b2, (short)0, acc2, false, false);
    acc3 = __builtin_amdgcn_wmma_f32_16x16x32_bf16(false, af, false, b3, (short)0, acc3, false, false);
  }

  // Epilogue. C layout: VGPR r -> row = r + (lane/16)*8 ; col = lane%16.
  int rbase = m0 + wm + (lane >> 4)*8;
  int c0 = n0 + wn + (lane & 15);
  int c1 = c0 + 16, c2 = c0 + 32, c3 = c0 + 48;
  #pragma unroll
  for (int r = 0; r < 8; ++r){
    int gr = rbase + r;
    float v0 = acc0[r], v1 = acc1[r], v2 = acc2[r], v3 = acc3[r];
    if (bias){ v0 += bias[c0]; v1 += bias[c1]; v2 += bias[c2]; v3 += bias[c3]; }
    if (addend){
      const float* ad = addend + (size_t)gr*N;
      v0 += ad[c0]; v1 += ad[c1]; v2 += ad[c2]; v3 += ad[c3];
    }
    if (relu){
      v0 = fmaxf(v0, 0.f); v1 = fmaxf(v1, 0.f);
      v2 = fmaxf(v2, 0.f); v3 = fmaxf(v3, 0.f);
    }
    if (Cf){
      float* cp = Cf + (size_t)gr*N;
      cp[c0] = v0; cp[c1] = v1; cp[c2] = v2; cp[c3] = v3;
    }
    if (Cb){
      bf16* cp = Cb + (size_t)gr*N;
      cp[c0] = (bf16)v0; cp[c1] = (bf16)v1; cp[c2] = (bf16)v2; cp[c3] = (bf16)v3;
    }
  }
}

// ---------------- gated accumulate of expert output ----------------
__global__ void dbm_expert_accum(const float* __restrict__ eo, const float* __restrict__ gw,
                                 const int* __restrict__ topi, int e, float* __restrict__ freq){
  int idx = blockIdx.x*blockDim.x + threadIdx.x;
  if (idx >= NB*NFR*256) return;
  int b = idx / (NFR*256);
  float w = 0.f;
  if (topi[b*2]   == e) w += gw[b*2];
  if (topi[b*2+1] == e) w += gw[b*2+1];
  if (w != 0.f) freq[idx] += w*eo[idx];
}

// ---------------- line conv1 (1->64, k7, pad3) + relu + maxpool4 ----------------
__global__ void dbm_line_conv1_pool(const float* __restrict__ x, const float* __restrict__ w,
                                    const float* __restrict__ bias, bf16* __restrict__ out){
  int idx = blockIdx.x*blockDim.x + threadIdx.x;       // NB*1024*64, BTC out
  if (idx >= NB*1024*64) return;
  int oc = idx & 63;
  int t1 = (idx >> 6) & 1023;
  int b  = idx >> 16;
  float m = -1e30f;
  for (int k4 = 0; k4 < 4; ++k4){
    int t = t1*4 + k4;
    float acc = bias[oc];
    #pragma unroll
    for (int kk = 0; kk < 7; ++kk){
      int tt = t + kk - 3;
      if (tt >= 0 && tt < 4096) acc += x[(size_t)b*4096 + tt]*w[oc*7 + kk];
    }
    m = fmaxf(m, acc);
  }
  out[idx] = (bf16)fmaxf(m, 0.f);
}

// ---------------- SE: mean over time (BTC bf16 -> fp32 (B,C)) ----------------
__global__ void dbm_se_mean(const bf16* __restrict__ x, float* __restrict__ y, int T, int C){
  int idx = blockIdx.x*blockDim.x + threadIdx.x;
  if (idx >= NB*C) return;
  int b = idx / C, c = idx % C;
  float s = 0.f;
  for (int t = 0; t < T; ++t) s += (float)x[(size_t)(b*T + t)*C + c];
  y[idx] = s / (float)T;
}

// ---------------- SE MLP: s = sigmoid(relu(y@w1.T)@w2.T) ----------------
__global__ void dbm_se_mlp(const float* __restrict__ y, const float* __restrict__ w1,
                           const float* __restrict__ w2, float* __restrict__ s,
                           int C, int CH){
  __shared__ float hid[64];
  int b = blockIdx.x, j = threadIdx.x;   // C threads
  if (j < CH){
    float a = 0.f;
    for (int k = 0; k < C; ++k) a += y[b*C + k]*w1[j*C + k];
    hid[j] = fmaxf(a, 0.f);
  }
  __syncthreads();
  float a = 0.f;
  for (int k = 0; k < CH; ++k) a += hid[k]*w2[j*CH + k];
  s[b*C + j] = sigf(a);
}

// ---------------- scale by SE + maxpool4 (BTC), bf16 and/or fp32 out ----------------
__global__ void dbm_scale_pool(const bf16* __restrict__ in, const float* __restrict__ s,
                               bf16* __restrict__ outb, float* __restrict__ outf,
                               int T, int C){
  int T4 = T >> 2;
  int idx = blockIdx.x*blockDim.x + threadIdx.x;
  if (idx >= NB*T4*C) return;
  int c  = idx % C;
  int t1 = (idx / C) % T4;
  int b  = idx / (C*T4);
  float m = -1e30f;
  for (int k = 0; k < 4; ++k) m = fmaxf(m, (float)in[(size_t)(b*T + t1*4 + k)*C + c]);
  float v = m * s[b*C + c];
  if (outb) outb[idx] = (bf16)v;
  if (outf) outf[idx] = v;
}

// ---------------- linear interp 64 -> 33 over time (BTC fp32) ----------------
__global__ void dbm_interp(const float* __restrict__ lf, float* __restrict__ out){
  int idx = blockIdx.x*blockDim.x + threadIdx.x;       // NB*33*256
  if (idx >= NB*NFR*256) return;
  int c = idx % 256;
  int i = (idx / 256) % NFR;
  int b = idx / (256*NFR);
  float scale = 64.0f/33.0f;
  float coord = fminf(fmaxf(((float)i + 0.5f)*scale - 0.5f, 0.f), 63.f);
  int lo = (int)floorf(coord);
  int hi = lo + 1; if (hi > 63) hi = 63;
  float w = coord - (float)lo;
  out[idx] = lf[(size_t)(b*64 + lo)*256 + c]*(1.f - w) + lf[(size_t)(b*64 + hi)*256 + c]*w;
}

// ---------------- build comb A matrix (t,b,f) bf16 ----------------
__global__ void dbm_comb(const float* __restrict__ freq, const float* __restrict__ line33,
                         bf16* __restrict__ Acomb){
  int idx = blockIdx.x*blockDim.x + threadIdx.x;       // 4224*512
  if (idx >= NFR*NB*512) return;
  int f = idx & 511;
  int r = idx >> 9;
  int t = r / NB, b = r % NB;
  float v = (f < 256) ? freq[(size_t)(b*NFR + t)*256 + f]
                      : line33[(size_t)(b*NFR + t)*256 + (f - 256)];
  Acomb[idx] = (bf16)v;
}

// ---------------- bias sum ----------------
__global__ void dbm_bsum(const float* __restrict__ a, const float* __restrict__ b,
                         float* __restrict__ o, int n){
  int i = blockIdx.x*blockDim.x + threadIdx.x;
  if (i < n) o[i] = a[i] + b[i];
}

// ---------------- LSTM cell (forward recurrence step) ----------------
__global__ void dbm_cell(const float* __restrict__ gates, float* __restrict__ c,
                         bf16* __restrict__ hb, float* __restrict__ last, int writeLast){
  int idx = blockIdx.x*blockDim.x + threadIdx.x;       // NB*256
  if (idx >= NB*256) return;
  int b = idx >> 8, hh = idx & 255;
  const float* g = gates + (size_t)b*1024;
  float ig = g[hh], fg = g[256 + hh], gg = g[512 + hh], og = g[768 + hh];
  float cv = sigf(fg)*c[idx] + sigf(ig)*tanhf(gg);
  float hv = sigf(og)*tanhf(cv);
  c[idx] = cv;
  hb[idx] = (bf16)hv;
  if (writeLast) last[(size_t)b*512 + hh] = hv;
}

// ---------------- backward LSTM: single step from zero state ----------------
__global__ void dbm_cell_bwd(const float* __restrict__ Xb, const float* __restrict__ bs,
                             float* __restrict__ last){
  int idx = blockIdx.x*blockDim.x + threadIdx.x;
  if (idx >= NB*256) return;
  int b = idx >> 8, hh = idx & 255;
  const float* g = Xb + (size_t)b*1024;
  float ig = g[hh] + bs[hh];
  float gg = g[512 + hh] + bs[512 + hh];
  float og = g[768 + hh] + bs[768 + hh];
  float cv = sigf(ig)*tanhf(gg);          // c starts at 0 -> forget term vanishes
  float hv = sigf(og)*tanhf(cv);
  last[(size_t)b*512 + 256 + hh] = hv;
}

// ---------------- final FFN 512 -> 256 -> 1 ----------------
__global__ void dbm_ffn(const float* __restrict__ last, const float* __restrict__ w1,
                        const float* __restrict__ b1, const float* __restrict__ w2,
                        const float* __restrict__ b2, float* __restrict__ out){
  __shared__ float hid[256];
  __shared__ float red[256];
  int b = blockIdx.x, j = threadIdx.x;    // 256 threads
  float a = b1[j];
  for (int k = 0; k < 512; ++k) a += last[(size_t)b*512 + k]*w1[j*512 + k];
  hid[j] = fmaxf(a, 0.f);
  __syncthreads();
  red[j] = hid[j]*w2[j];
  __syncthreads();
  for (int s = 128; s > 0; s >>= 1){
    if (j < s) red[j] += red[j + s];
    __syncthreads();
  }
  if (j == 0) out[b] = red[0] + b2[0];
}

// =====================================================================
extern "C" void kernel_launch(void* const* d_in, const int* in_sizes, int n_in,
                              void* d_out, int out_size, void* d_ws, size_t ws_size,
                              hipStream_t stream){
  const float* x_cont = (const float*)d_in[0];
  const float* x_norm = (const float*)d_in[1];
  const float* gate_w1 = (const float*)d_in[2];
  const float* gate_b1 = (const float*)d_in[3];
  const float* gate_w2 = (const float*)d_in[4];
  const float* gate_b2 = (const float*)d_in[5];
  const float* exp_w1 = (const float*)d_in[6];
  const float* exp_b1 = (const float*)d_in[7];
  const float* exp_w2 = (const float*)d_in[8];
  const float* exp_b2 = (const float*)d_in[9];
  const float* lw1 = (const float*)d_in[10];
  const float* lb1 = (const float*)d_in[11];
  const float* lw2 = (const float*)d_in[12];
  const float* lb2 = (const float*)d_in[13];
  const float* se2_w1 = (const float*)d_in[14];
  const float* se2_w2 = (const float*)d_in[15];
  const float* lw3 = (const float*)d_in[16];
  const float* lb3 = (const float*)d_in[17];
  const float* se3_w1 = (const float*)d_in[18];
  const float* se3_w2 = (const float*)d_in[19];
  const float* wih_f = (const float*)d_in[20];
  const float* whh_f = (const float*)d_in[21];
  const float* bih_f = (const float*)d_in[22];
  const float* bhh_f = (const float*)d_in[23];
  const float* wih_b = (const float*)d_in[24];
  const float* bih_b = (const float*)d_in[26];
  const float* bhh_b = (const float*)d_in[27];
  const float* ffn_w1 = (const float*)d_in[28];
  const float* ffn_b1 = (const float*)d_in[29];
  const float* ffn_w2 = (const float*)d_in[30];
  const float* ffn_b2 = (const float*)d_in[31];
  float* out = (float*)d_out;

  // ---- workspace carve ----
  char* p = (char*)d_ws;
  auto alloc = [&](size_t bytes) -> void* {
    void* r = (void*)p;
    p += (bytes + 255) & ~(size_t)255;
    return r;
  };
  const int MROWS = NB*NFR;                     // 4224
  float* magf   = (float*)alloc((size_t)MROWS*NBIN*4);
  bf16*  magb   = (bf16*) alloc((size_t)MROWS*NBIN*2);
  float* pooled = (float*)alloc((size_t)NB*NBIN*4);
  float* gwbuf  = (float*)alloc(NB*2*4);
  int*   topib  = (int*)  alloc(NB*2*4);
  bf16*  expB1  = (bf16*) alloc((size_t)672*256*2);
  bf16*  expB2  = (bf16*) alloc((size_t)768*256*2);
  bf16*  h1     = (bf16*) alloc((size_t)MROWS*256*2);
  float* eo     = (float*)alloc((size_t)MROWS*256*4);
  float* freq   = (float*)alloc((size_t)MROWS*256*4);
  bf16*  pooled1= (bf16*) alloc((size_t)NB*1024*64*2);
  bf16*  lineB2 = (bf16*) alloc((size_t)320*128*2);
  bf16*  h2     = (bf16*) alloc((size_t)NB*1024*128*2);
  float* ym2    = (float*)alloc((size_t)NB*128*4);
  float* s2     = (float*)alloc((size_t)NB*128*4);
  bf16*  pooled2= (bf16*) alloc((size_t)NB*256*128*2);
  bf16*  lineB3 = (bf16*) alloc((size_t)384*256*2);
  bf16*  h3     = (bf16*) alloc((size_t)NB*256*256*2);
  float* ym3    = (float*)alloc((size_t)NB*256*4);
  float* s3     = (float*)alloc((size_t)NB*256*4);
  float* linef  = (float*)alloc((size_t)NB*64*256*4);
  float* line33 = (float*)alloc((size_t)NB*NFR*256*4);
  bf16*  Acomb  = (bf16*) alloc((size_t)MROWS*512*2);
  bf16*  WihF   = (bf16*) alloc((size_t)512*1024*2);
  bf16*  WihB   = (bf16*) alloc((size_t)512*1024*2);
  bf16*  WhhF   = (bf16*) alloc((size_t)256*1024*2);
  float* Xf     = (float*)alloc((size_t)MROWS*1024*4);
  float* XbL    = (float*)alloc((size_t)NB*1024*4);
  float* bsF    = (float*)alloc(1024*4);
  float* bsB    = (float*)alloc(1024*4);
  float* gates  = (float*)alloc((size_t)NB*1024*4);
  bf16*  hbf    = (bf16*) alloc((size_t)NB*256*2);
  float* cbuf   = (float*)alloc((size_t)NB*256*4);
  float* lastb  = (float*)alloc((size_t)NB*512*4);
  (void)ws_size; (void)in_sizes; (void)n_in; (void)out_size;

  // ---- 1. STFT + pooled + gate ----
  dbm_stft<<<NB*NFR, 256, 0, stream>>>(x_cont, magf, magb);
  dbm_pool_mean<<<(NB*NBIN + 255)/256, 256, 0, stream>>>(magf, pooled);
  dbm_gate<<<NB, 128, 0, stream>>>(pooled, gate_w1, gate_b1, gate_w2, gate_b2, gwbuf, topib);

  // ---- 2. MoE experts (implicit-im2col WMMA GEMMs), gated accumulation ----
  hipMemsetAsync(freq, 0, (size_t)MROWS*256*4, stream);
  for (int e = 0; e < 8; ++e){
    dbm_pack_wB<<<(672*256 + 255)/256, 256, 0, stream>>>(exp_w1 + (size_t)e*256*645, expB1, 256, 645, 672);
    dbm_pack_wB<<<(768*256 + 255)/256, 256, 0, stream>>>(exp_w2 + (size_t)e*256*768, expB2, 256, 768, 768);
    dbm_gemm<1, NFR, NBIN, 5, 2><<<dim3(MROWS/32, 256/128), 128, 0, stream>>>(
        magb, expB1, exp_b1 + e*256, nullptr, nullptr, h1, MROWS, 256, 672, 1);
    dbm_gemm<1, NFR, 256, 3, 1><<<dim3(MROWS/32, 256/128), 128, 0, stream>>>(
        h1, expB2, exp_b2 + e*256, nullptr, eo, nullptr, MROWS, 256, 768, 1);
    dbm_expert_accum<<<(MROWS*256 + 255)/256, 256, 0, stream>>>(eo, gwbuf, topib, e, freq);
  }

  // ---- 3. line branch ----
  dbm_line_conv1_pool<<<(NB*1024*64 + 255)/256, 256, 0, stream>>>(x_norm, lw1, lb1, pooled1);
  dbm_pack_wB<<<(320*128 + 255)/256, 256, 0, stream>>>(lw2, lineB2, 128, 320, 320);
  dbm_gemm<1, 1024, 64, 5, 2><<<dim3(NB*1024/32, 1), 128, 0, stream>>>(
      pooled1, lineB2, lb2, nullptr, nullptr, h2, NB*1024, 128, 320, 1);
  dbm_se_mean<<<(NB*128 + 255)/256, 256, 0, stream>>>(h2, ym2, 1024, 128);
  dbm_se_mlp<<<NB, 128, 0, stream>>>(ym2, se2_w1, se2_w2, s2, 128, 32);
  dbm_scale_pool<<<(NB*256*128 + 255)/256, 256, 0, stream>>>(h2, s2, pooled2, nullptr, 1024, 128);
  dbm_pack_wB<<<(384*256 + 255)/256, 256, 0, stream>>>(lw3, lineB3, 256, 384, 384);
  dbm_gemm<1, 256, 128, 3, 1><<<dim3(NB*256/32, 256/128), 128, 0, stream>>>(
      pooled2, lineB3, lb3, nullptr, nullptr, h3, NB*256, 256, 384, 1);
  dbm_se_mean<<<(NB*256 + 255)/256, 256, 0, stream>>>(h3, ym3, 256, 256);
  dbm_se_mlp<<<NB, 256, 0, stream>>>(ym3, se3_w1, se3_w2, s3, 256, 64);
  dbm_scale_pool<<<(NB*64*256 + 255)/256, 256, 0, stream>>>(h3, s3, nullptr, linef, 256, 256);
  dbm_interp<<<(NB*NFR*256 + 255)/256, 256, 0, stream>>>(linef, line33);

  // ---- 4. comb + hoisted input projections ----
  dbm_comb<<<(MROWS*512 + 255)/256, 256, 0, stream>>>(freq, line33, Acomb);
  dbm_pack_wB<<<(512*1024 + 255)/256, 256, 0, stream>>>(wih_f, WihF, 1024, 512, 512);
  dbm_pack_wB<<<(512*1024 + 255)/256, 256, 0, stream>>>(wih_b, WihB, 1024, 512, 512);
  dbm_pack_wB<<<(256*1024 + 255)/256, 256, 0, stream>>>(whh_f, WhhF, 1024, 256, 256);
  dbm_gemm<0, 1, 1, 1, 0><<<dim3(MROWS/32, 1024/128), 128, 0, stream>>>(
      Acomb, WihF, nullptr, nullptr, Xf, nullptr, MROWS, 1024, 512, 0);
  // backward LSTM only needs its first step (== hb[-1] after re-reversal): t = 32
  dbm_gemm<0, 1, 1, 1, 0><<<dim3(NB/32, 1024/128), 128, 0, stream>>>(
      Acomb + (size_t)32*NB*512, WihB, nullptr, nullptr, XbL, nullptr, NB, 1024, 512, 0);
  dbm_bsum<<<4, 256, 0, stream>>>(bih_f, bhh_f, bsF, 1024);
  dbm_bsum<<<4, 256, 0, stream>>>(bih_b, bhh_b, bsB, 1024);

  // ---- 5. forward LSTM recurrence (33 steps) ----
  hipMemsetAsync(hbf,  0, (size_t)NB*256*2, stream);
  hipMemsetAsync(cbuf, 0, (size_t)NB*256*4, stream);
  for (int t = 0; t < NFR; ++t){
    dbm_gemm<0, 1, 1, 1, 0><<<dim3(NB/32, 1024/128), 128, 0, stream>>>(
        hbf, WhhF, bsF, Xf + (size_t)t*NB*1024, gates, nullptr, NB, 1024, 256, 0);
    dbm_cell<<<(NB*256 + 255)/256, 256, 0, stream>>>(gates, cbuf, hbf, lastb, (t == NFR-1) ? 1 : 0);
  }
  dbm_cell_bwd<<<(NB*256 + 255)/256, 256, 0, stream>>>(XbL, bsB, lastb);

  // ---- 6. FFN head ----
  dbm_ffn<<<NB, 256, 0, stream>>>(lastb, ffn_w1, ffn_b1, ffn_w2, ffn_b2, out);
}
